// SelfAttentionLSA_42047729827989
// MI455X (gfx1250) — compile-verified, matrix operand order checked by reading, not run on hardware
//
#include <hip/hip_runtime.h>

// ---------------------------------------------------------------------------
// Types for CDNA5 WMMA (wave32): v_wmma_f32_16x16x32_bf16
// ---------------------------------------------------------------------------
typedef __bf16 bf16;
typedef __attribute__((ext_vector_type(4)))  __bf16 v4bf;
typedef __attribute__((ext_vector_type(8)))  __bf16 v8bf;
typedef __attribute__((ext_vector_type(16))) __bf16 v16bf;
typedef __attribute__((ext_vector_type(8)))  float  v8f;
typedef __attribute__((ext_vector_type(4)))  int    v4i;   // 128-bit async-copy payload

#define BM 128
#define BN 128
#define BK 32
#define LDSS 48   // 32 + 16 pad (keeps 16B alignment, staggers banks)

#define AS1 __attribute__((address_space(1)))
#define AS3 __attribute__((address_space(3)))

#if defined(__has_builtin)
#if __has_builtin(__builtin_amdgcn_global_load_async_to_lds_b128)
#define LSA_HAVE_ASYNC 1
#endif
#endif

// ---------------------------------------------------------------------------
// fp32 -> bf16 conversion, 4 elements / thread
// ---------------------------------------------------------------------------
__global__ __launch_bounds__(256)
void lsa_cvt_f32_bf16_x4(const float* __restrict__ src, bf16* __restrict__ dst, int n4) {
    int i = blockIdx.x * 256 + threadIdx.x;
    if (i >= n4) return;
    float4 v = ((const float4*)src)[i];
    v4bf o;
    o[0] = (bf16)v.x; o[1] = (bf16)v.y; o[2] = (bf16)v.z; o[3] = (bf16)v.w;
    ((v4bf*)dst)[i] = o;
}

// ---------------------------------------------------------------------------
// CDNA5 async global->LDS staging of one 128x32 bf16 tile pair.
// ASYNCcnt path (GLOBAL_LOAD_ASYNC_TO_LDS_B128) when available; otherwise
// classic reg-staged copy (global_load_b128 -> ds_store_b128).
// ---------------------------------------------------------------------------
__device__ __forceinline__ void lsa_stage(const bf16* __restrict__ Ag,
                                          const bf16* __restrict__ Bg,
                                          int lda, int ldb, int kt,
                                          bf16 (*As)[LDSS], bf16 (*Bs)[LDSS],
                                          int tid)
{
#pragma unroll
    for (int it = 0; it < 2; ++it) {
        int c   = tid + it * 256;          // 0..511 chunk id
        int row = c >> 2;                  // 4 x 16B chunks per 32-elem row
        int col = (c & 3) * 8;
#if defined(LSA_HAVE_ASYNC)
        __builtin_amdgcn_global_load_async_to_lds_b128(
            (v4i*)(uintptr_t)(Ag + (long)row * lda + kt + col),
            (AS3 v4i*)(unsigned)(uintptr_t)(&As[row][col]),
            0, 0);
        __builtin_amdgcn_global_load_async_to_lds_b128(
            (v4i*)(uintptr_t)(Bg + (long)row * ldb + kt + col),
            (AS3 v4i*)(unsigned)(uintptr_t)(&Bs[row][col]),
            0, 0);
#else
        *(uint4*)(&As[row][col]) = *(const uint4*)(Ag + (long)row * lda + kt + col);
        *(uint4*)(&Bs[row][col]) = *(const uint4*)(Bg + (long)row * ldb + kt + col);
#endif
    }
}

__device__ __forceinline__ void lsa_wait_async() {
#if defined(LSA_HAVE_ASYNC)
#if __has_builtin(__builtin_amdgcn_s_wait_asynccnt)
    __builtin_amdgcn_s_wait_asynccnt(0);
#else
    asm volatile("s_wait_asynccnt 0x0" ::: "memory");
#endif
#endif
}

// ---------------------------------------------------------------------------
// Load a 16x32 bf16 fragment from LDS in the ISA's A/B register layout:
//   lanes 0-15 : row = lane,    K = 0..7  (VGPR0-3),  K = 16..23 (VGPR4-7)
//   lanes16-31 : row = lane-16, K = 8..15 (VGPR0-3),  K = 24..31 (VGPR4-7)
// ---------------------------------------------------------------------------
__device__ __forceinline__ v16bf lsa_frag(const bf16* __restrict__ rowptr, int kb) {
    v8bf lo = *(const v8bf*)(rowptr + kb);
    v8bf hi = *(const v8bf*)(rowptr + kb + 16);
    return __builtin_shufflevector(lo, hi, 0,1,2,3,4,5,6,7,8,9,10,11,12,13,14,15);
}

// ---------------------------------------------------------------------------
// Tiled NT GEMM:  C[M,N] = scale * A[M,K] (row) x B[N,K] (row)^T, bf16 in,
// f32 accumulate via v_wmma_f32_16x16x32_bf16.
// Block: 128x128, 256 threads = 8 waves (4x2), each wave a 32x64 sub-tile.
// Pipelined: async-stage tile t+1 into LDS while tile t's WMMAs run.
// MODE 0: f32 out   MODE 1: bf16 out   MODE 2: bf16 out, transposed+batched
// ---------------------------------------------------------------------------
template <int MODE>
__global__ __launch_bounds__(256)
void lsa_gemm_nt_wmma(const bf16* __restrict__ A, const bf16* __restrict__ B,
                      void* __restrict__ Cv,
                      int K, int lda, int ldb, int ldc,
                      long strideA, long strideB, long strideC,
                      const float* __restrict__ temp_ptr,   // scale = 1/temp (nullptr -> 1)
                      int trl2, long tBatchStride, int ldcT)
{
    __shared__ alignas(16) bf16 As[BM][LDSS];
    __shared__ alignas(16) bf16 Bs[BN][LDSS];

    const int tid   = threadIdx.x;
    const int lane  = tid & 31;
    const int wave  = tid >> 5;
    const int waveM = wave >> 1;          // 0..3 -> 32-row strip
    const int waveN = wave & 1;           // 0..1 -> 64-col strip
    const int z     = blockIdx.z;

    const bf16* Ag = A + (long)z * strideA + (long)blockIdx.y * BM * lda;
    const bf16* Bg = B + (long)z * strideB + (long)blockIdx.x * BN * ldb;

    const int mrow = lane & 15;
    const int kb   = (lane >> 4) * 8;

    v8f acc[2][4];
#pragma unroll
    for (int i = 0; i < 2; ++i)
#pragma unroll
        for (int j = 0; j < 4; ++j) acc[i][j] = {};

    // Prologue: stage tile 0
    lsa_stage(Ag, Bg, lda, ldb, 0, As, Bs, tid);

    for (int kt = 0; kt < K; kt += BK) {
        lsa_wait_async();
        __syncthreads();                  // tile kt visible to all waves

        v16bf af[2], bfr[4];
#pragma unroll
        for (int i = 0; i < 2; ++i)
            af[i] = lsa_frag(&As[waveM * 32 + i * 16 + mrow][0], kb);
#pragma unroll
        for (int j = 0; j < 4; ++j)
            bfr[j] = lsa_frag(&Bs[waveN * 64 + j * 16 + mrow][0], kb);

        __syncthreads();                  // all LDS reads retired -> buffer free

        if (kt + BK < K)                  // overlap next tile's DMA with WMMAs
            lsa_stage(Ag, Bg, lda, ldb, kt + BK, As, Bs, tid);
        if (kt + 2 * BK < K) {            // warm GL2 for the tile after next
            int row = tid >> 2, col = (tid & 3) * 8;
            __builtin_prefetch(Ag + (long)row * lda + kt + 2 * BK + col, 0, 1);
            __builtin_prefetch(Bg + (long)row * ldb + kt + 2 * BK + col, 0, 1);
        }

#pragma unroll
        for (int i = 0; i < 2; ++i)
#pragma unroll
            for (int j = 0; j < 4; ++j)
                acc[i][j] = __builtin_amdgcn_wmma_f32_16x16x32_bf16(
                    false, af[i], false, bfr[j], (short)0, acc[i][j], false, false);
    }

    const float s = temp_ptr ? (1.0f / temp_ptr[0]) : 1.0f;

    // C/D layout: lanes 0-15 -> N=lane, M=vgpr r; lanes 16-31 -> N=lane-16, M=8+r
    const int laneM   = (lane >> 4) * 8;
    const int laneN   = lane & 15;
    const int rowBase = blockIdx.y * BM + waveM * 32;
    const int colBase = blockIdx.x * BN + waveN * 64;

#pragma unroll
    for (int i = 0; i < 2; ++i) {
#pragma unroll
        for (int j = 0; j < 4; ++j) {
#pragma unroll
            for (int r = 0; r < 8; ++r) {
                int   row = rowBase + i * 16 + laneM + r;
                int   col = colBase + j * 16 + laneN;
                float v   = acc[i][j][r] * s;
                if constexpr (MODE == 0) {
                    float* C = (float*)Cv + (long)z * strideC;
                    C[(long)row * ldc + col] = v;
                } else if constexpr (MODE == 1) {
                    bf16* C = (bf16*)Cv + (long)z * strideC;
                    C[(long)row * ldc + col] = (bf16)v;
                } else {
                    // transposed + batched: out[b][col][row%rows], b = row>>trl2
                    bf16* C  = (bf16*)Cv;
                    int   bb = row >> trl2;
                    int   rm = row & ((1 << trl2) - 1);
                    C[(long)bb * tBatchStride + (long)col * ldcT + rm] = (bf16)v;
                }
            }
        }
    }
}

// ---------------------------------------------------------------------------
// Column-wise softmax (reference softmaxes over axis=1 == the query axis):
// for each column m of S[b] (1024x1024 f32), online max/sum over rows n,
// write normalized probabilities as bf16.
// ---------------------------------------------------------------------------
__global__ __launch_bounds__(256)
void lsa_col_softmax_bf16(const float* __restrict__ S, bf16* __restrict__ P, int N) {
    const int  b   = blockIdx.y;
    const int  m   = blockIdx.x * 256 + threadIdx.x;
    const float* col = S + (long)b * N * N + m;

    float mx = -3.402823466e38f, sum = 0.0f;
    for (int n = 0; n < N; ++n) {
        float v = col[(long)n * N];
        if (v > mx) { sum *= __expf(mx - v); mx = v; }
        sum += __expf(v - mx);
    }
    const float inv = 1.0f / sum;

    bf16* pc = P + (long)b * N * N + m;
    for (int n = 0; n < N; ++n) {
        float v = col[(long)n * N];
        pc[(long)n * N] = (bf16)(__expf(v - mx) * inv);
    }
}

// ---------------------------------------------------------------------------
// Launcher
// ---------------------------------------------------------------------------
extern "C" void kernel_launch(void* const* d_in, const int* in_sizes, int n_in,
                              void* d_out, int out_size, void* d_ws, size_t ws_size,
                              hipStream_t stream) {
    const float* x    = (const float*)d_in[0];
    const float* Wq   = (const float*)d_in[1];
    const float* Wk   = (const float*)d_in[2];
    const float* Wv   = (const float*)d_in[3];
    const float* temp = (const float*)d_in[4];

    const int  Bz = 16, N = 1024, C = 768;
    const long MX = (long)Bz * N;            // 16384 rows of x

    // Workspace carve-up (256B aligned)
    char*  ws  = (char*)d_ws;
    size_t off = 0;
    auto carve = [&](size_t bytes) -> void* {
        void* p = ws + off;
        off += (bytes + 255) & ~(size_t)255;
        return p;
    };
    bf16*  xb   = (bf16*)carve((size_t)MX * C * 2);
    bf16*  wqb  = (bf16*)carve((size_t)C * C * 2);
    bf16*  wkb  = (bf16*)carve((size_t)C * C * 2);
    bf16*  wvb  = (bf16*)carve((size_t)C * C * 2);
    bf16*  qb   = (bf16*)carve((size_t)MX * C * 2);
    bf16*  kbuf = (bf16*)carve((size_t)MX * C * 2);
    bf16*  vtb  = (bf16*)carve((size_t)MX * C * 2);         // V transposed per batch: [b][d][m]
    float* attn = (float*)carve((size_t)Bz * N * N * 4);
    bf16*  pb   = (bf16*)carve((size_t)Bz * N * N * 2);
    (void)ws_size; (void)in_sizes; (void)n_in; (void)out_size;

    // 1) fp32 -> bf16
    {
        int n4 = (int)(MX * C / 4);
        lsa_cvt_f32_bf16_x4<<<(n4 + 255) / 256, 256, 0, stream>>>(x, xb, n4);
        int w4 = C * C / 4;
        lsa_cvt_f32_bf16_x4<<<(w4 + 255) / 256, 256, 0, stream>>>(Wq, wqb, w4);
        lsa_cvt_f32_bf16_x4<<<(w4 + 255) / 256, 256, 0, stream>>>(Wk, wkb, w4);
        lsa_cvt_f32_bf16_x4<<<(w4 + 255) / 256, 256, 0, stream>>>(Wv, wvb, w4);
    }

    dim3 blk(256);

    // 2) Projections: [16384x768] @ [768x768]^T
    dim3 gp(C / BN, MX / BM, 1);
    lsa_gemm_nt_wmma<1><<<gp, blk, 0, stream>>>(xb, wqb, qb,   C, C, C, C, 0, 0, 0, nullptr, 0, 0, 0);
    lsa_gemm_nt_wmma<1><<<gp, blk, 0, stream>>>(xb, wkb, kbuf, C, C, C, C, 0, 0, 0, nullptr, 0, 0, 0);
    lsa_gemm_nt_wmma<2><<<gp, blk, 0, stream>>>(xb, wvb, vtb,  C, C, C, 0, 0, 0, 0, nullptr,
                                                10 /*log2(N)*/, (long)C * N, N);

    // 3) Scores: per batch Q[1024x768] @ K[1024x768]^T, scaled by 1/temperature
    dim3 ga(N / BN, N / BM, Bz);
    lsa_gemm_nt_wmma<0><<<ga, blk, 0, stream>>>(qb, kbuf, attn, C, C, C, N,
                                                (long)N * C, (long)N * C, (long)N * N,
                                                temp, 0, 0, 0);

    // 4) Column-wise softmax -> bf16 probabilities
    dim3 gs(N / 256, Bz);
    lsa_col_softmax_bf16<<<gs, 256, 0, stream>>>(attn, pb, N);

    // 5) Output: per batch P[1024x1024] @ Vt[768x1024]^T -> f32 d_out
    dim3 go(C / BN, N / BM, Bz);
    lsa_gemm_nt_wmma<0><<<go, blk, 0, stream>>>(pb, vtb, (float*)d_out, N, N, N, C,
                                                (long)N * N, (long)C * N, (long)N * C,
                                                nullptr, 0, 0, 0);
}
